// RelativePositionTransformerEncoder_37280316129771
// MI455X (gfx1250) — matrix operand order repaired and checked by reading, not compiled, not run on hardware
//
#include <hip/hip_runtime.h>

// ---------------------------------------------------------------------------
// MI455X (gfx1250) BF16-WMMA transformer encoder.
// GEMMs: v_wmma_f32_16x16x32_bf16 with f32 accumulation.
// Tile movement: Tensor Data Mover (tensor_load_to_lds) with double buffering,
// tracked by TENSORcnt (s_wait_tensorcnt). Fragments read via ds_load_b128.
// ---------------------------------------------------------------------------

typedef __attribute__((ext_vector_type(16))) __bf16 v16bf;
typedef __attribute__((ext_vector_type(8)))  float  v8f;
typedef __attribute__((ext_vector_type(4)))  unsigned int u32x4;
typedef __attribute__((ext_vector_type(8)))  int i32x8;
typedef __attribute__((ext_vector_type(4)))  int i32x4;

#define L_ 6
#define H_ 8
#define D_ 1024
#define FF_ 2048
#define E_ 128
#define S_ 512
#define B_ 16
#define MAXLEN_ 1024
#define NROWS_ 8192   // B_*S_

#if defined(__HIP_DEVICE_COMPILE__) && __has_builtin(__builtin_amdgcn_tensor_load_to_lds)
#define HAVE_TDM 1
#else
#define HAVE_TDM 0
#endif

#if HAVE_TDM
// Issue one TDM descriptor: load a [128 rows x 32 elems] bf16 tile whose rows
// are strided by `stride` elements in global memory into contiguous LDS rows
// of 64B at lds_off. D# layout per CDNA5 ISA 8.3/8.4.
__device__ __forceinline__ void tdm_load_tile(unsigned lds_off, const __bf16* gptr,
                                              int stride_elems)
{
    unsigned long long ga = (unsigned long long)(uintptr_t)gptr;
    u32x4 g0;
    g0[0] = 1u;                                            // count=1, user mode
    g0[1] = lds_off;                                       // lds_addr (bytes)
    g0[2] = (unsigned)(ga & 0xFFFFFFFFu);                  // global_addr[31:0]
    g0[3] = (unsigned)((ga >> 32) & 0x01FFFFFFu)           // global_addr[56:32]
            | 0x80000000u;                                 // type=2 ("image")
    i32x8 g1;
    g1[0] = 0x00010000;          // workgroup_mask=0, data_size=1 (2 bytes)
    g1[1] = (int)(32u << 16);    // tensor_dim0[15:0]=32
    g1[2] = (int)(128u << 16);   // tensor_dim0[31:16]=0 | tensor_dim1[15:0]=128
    g1[3] = (int)(32u << 16);    // tensor_dim1[31:16]=0 | tile_dim0=32
    g1[4] = 128;                 // tile_dim1=128, tile_dim2=0
    g1[5] = stride_elems;        // tensor_dim0_stride[31:0]
    g1[6] = 0;                   // stride hi, tensor_dim1_stride lo
    g1[7] = 0;
    i32x4 z4 = {0, 0, 0, 0};
#if __clang_major__ >= 23
    i32x8 z8 = {0, 0, 0, 0, 0, 0, 0, 0};
    __builtin_amdgcn_tensor_load_to_lds(g0, g1, z4, z4, z8, 0);
#else
    __builtin_amdgcn_tensor_load_to_lds(g0, g1, z4, z4, 0);
#endif
}
#endif

// ---------------------------------------------------------------------------
// Batched WMMA GEMM: C[z] (+)= A[z] @ BT[z]^T  (bf16 in, f32 out)
// A: [M,K] row-major (lda); BT: [N,K] row-major (ldb)  -> C: [M,N] (ldc).
// Block tile 128x128, K-step 32, 8 waves (wave32), wave tile 32x64.
// ---------------------------------------------------------------------------
template<bool ACCUM, bool BIAS, bool RELU>
__global__ __launch_bounds__(256)
void gemm_wmma_bf16(const __bf16* __restrict__ A, const __bf16* __restrict__ BT,
                    float* __restrict__ C, __bf16* __restrict__ Cb,
                    const float* __restrict__ bias,
                    int K, int lda, int ldb, int ldc,
                    long long sAz, long long sBz, long long sCz)
{
#if HAVE_TDM
    __shared__ unsigned short As[2][128 * 32];   // [m][k], 64B rows
    __shared__ unsigned short Bs[2][128 * 32];   // [n][k], 64B rows
#else
    __shared__ unsigned short As[1][128 * 32];
    __shared__ unsigned short Bs[1][128 * 32];
#endif

    const int tid  = threadIdx.x;
    const int lane = tid & 31;
    const int wv   = tid >> 5;
    const int wm   = wv & 3;          // M subtile (x32)
    const int wn   = wv >> 2;         // N subtile (x64)
    const int hi   = lane >> 4;
    const int l15  = lane & 15;
    const int z    = blockIdx.z;
    const int bm   = blockIdx.y * 128;
    const int bn   = blockIdx.x * 128;

    A  += (long long)z * sAz;
    BT += (long long)z * sBz;
    C  += (long long)z * sCz;
    if (Cb) Cb += (long long)z * sCz;

    const __bf16* Abase = A  + (long long)bm * lda;   // tile row origin
    const __bf16* Bbase = BT + (long long)bn * ldb;

    v8f acc[2][4];
#pragma unroll
    for (int i = 0; i < 2; ++i)
#pragma unroll
        for (int j = 0; j < 4; ++j)
#pragma unroll
            for (int e = 0; e < 8; ++e) acc[i][j][e] = 0.0f;

    const int nk = K >> 5;

#if HAVE_TDM
    const unsigned As_off = (unsigned)(uintptr_t)&As[0][0];
    const unsigned Bs_off = (unsigned)(uintptr_t)&Bs[0][0];
    // Prologue: tiles for iteration 0 into buffer 0.
    if (tid < 32) {
        tdm_load_tile(As_off, Abase, lda);
        tdm_load_tile(Bs_off, Bbase, ldb);
    }
#endif

    for (int i = 0; i < nk; ++i) {
#if HAVE_TDM
        const int cur = i & 1;
        if (tid < 32) {
            if (i + 1 < nk) {   // prefetch next tiles into other buffer
                const unsigned boff = (unsigned)((i + 1) & 1) * (128 * 32 * 2);
                tdm_load_tile(As_off + boff, Abase + (i + 1) * 32, lda);
                tdm_load_tile(Bs_off + boff, Bbase + (i + 1) * 32, ldb);
                // In-order completion: <=2 outstanding => current tiles done.
                __builtin_amdgcn_s_wait_tensorcnt(2);
            } else {
                __builtin_amdgcn_s_wait_tensorcnt(0);
            }
        }
        __syncthreads();
        // LDS arrays are module-scope globals; a memory clobber tells the
        // compiler they may have been written (by the TDM), so the ds_load
        // fragment reads below cannot be folded away. Emits no instructions.
        asm volatile("" ::: "memory");
        const unsigned short* Asl = &As[cur][0];
        const unsigned short* Bsl = &Bs[cur][0];
#else
        // Manual staging: both tiles are [row][32] with aligned strides.
        {
            const int idx = tid * 4;
            const int r   = idx >> 5;
            const int c   = idx & 31;
#pragma unroll
            for (int it = 0; it < 4; ++it) {
                const int row = r + it * 32;
                *(uint2*)&As[0][row * 32 + c] =
                    *(const uint2*)(Abase + (long long)row * lda + (i * 32 + c));
                *(uint2*)&Bs[0][row * 32 + c] =
                    *(const uint2*)(Bbase + (long long)row * ldb + (i * 32 + c));
            }
        }
        __syncthreads();
        const unsigned short* Asl = &As[0][0];
        const unsigned short* Bsl = &Bs[0][0];
#endif

        union Frag { unsigned int u[8]; v16bf v; };
        Frag af[2];
#pragma unroll
        for (int mf = 0; mf < 2; ++mf) {
            const int row = wm * 32 + mf * 16 + l15;
            const unsigned int* Au = (const unsigned int*)&Asl[row * 32];
#pragma unroll
            for (int r = 0; r < 8; ++r) {
                const int kk = hi * 16 + r * 2;
                af[mf].u[r] = Au[kk >> 1];
            }
        }
#pragma unroll
        for (int nf = 0; nf < 4; ++nf) {
            const int n = wn * 64 + nf * 16 + l15;
            const unsigned int* Bu = (const unsigned int*)&Bsl[n * 32];
            Frag bfr;
#pragma unroll
            for (int r = 0; r < 8; ++r) bfr.u[r] = Bu[r + hi * 8];
#pragma unroll
            for (int mf = 0; mf < 2; ++mf) {
                acc[mf][nf] = __builtin_amdgcn_wmma_f32_16x16x32_bf16(
                    false, af[mf].v, false, bfr.v, (short)0, acc[mf][nf], false, false);
            }
        }
        __syncthreads();
    }

    // ---- epilogue ----
#pragma unroll
    for (int mf = 0; mf < 2; ++mf) {
#pragma unroll
        for (int nf = 0; nf < 4; ++nf) {
            const int col = bn + wn * 64 + nf * 16 + l15;
#pragma unroll
            for (int r = 0; r < 8; ++r) {
                const int row = bm + wm * 32 + mf * 16 + hi * 8 + r;
                const long long off = (long long)row * ldc + col;
                float val = acc[mf][nf][r];
                if (ACCUM) val += C[off];
                if (BIAS)  val += bias[col];
                if (RELU)  val = val > 0.0f ? val : 0.0f;
                C[off] = val;
                if (Cb) Cb[off] = (__bf16)val;
            }
        }
    }
}

// ---------------------------------------------------------------------------
// Elementwise helper kernels
// ---------------------------------------------------------------------------
__global__ void f32_to_bf16_k(const float* __restrict__ in, __bf16* __restrict__ out,
                              long long n)
{
    long long i = (long long)blockIdx.x * 256 + threadIdx.x;
    if (i < n) out[i] = (__bf16)in[i];
}

// in: [K,N] f32 row-major  ->  out: [N,K] bf16 row-major (transposed operand)
__global__ void transpose_bf16_k(const float* __restrict__ in, __bf16* __restrict__ out,
                                 int K, int N)
{
    long long i = (long long)blockIdx.x * 256 + threadIdx.x;   // over K*N
    int k = (int)(i / N);
    int n = (int)(i - (long long)k * N);
    out[(long long)n * K + k] = (__bf16)in[i];
}

// q,k,v f32 [B,S,H,E] -> Qbu/Qbv/Kc bf16 [B,H,S,E],  VT bf16 [B,H,E,S]
__global__ void pack_qkv_k(const float* __restrict__ q, const float* __restrict__ k,
                           const float* __restrict__ v,
                           const float* __restrict__ bu, const float* __restrict__ bv,
                           __bf16* __restrict__ Qbu, __bf16* __restrict__ Qbv,
                           __bf16* __restrict__ Kc, __bf16* __restrict__ VT)
{
    long long i = (long long)blockIdx.x * 256 + threadIdx.x;   // B*S*H*E
    int e = (int)(i & 127);
    int h = (int)((i >> 7) & 7);
    int s = (int)((i >> 10) & 511);
    int b = (int)(i >> 19);
    float qf = q[i], kf = k[i], vf = v[i];
    long long zo = (((long long)(b * H_ + h) * S_) + s) * E_ + e;
    Qbu[zo] = (__bf16)(qf + bu[h * E_ + e]);
    Qbv[zo] = (__bf16)(qf + bv[h * E_ + e]);
    Kc[zo]  = (__bf16)kf;
    VT[((long long)(b * H_ + h) * E_ + e) * S_ + s] = (__bf16)vf;
}

// oc f32 [B,H,S,E] -> ob bf16 [B,S,H*E]
__global__ void permute_o_k(const float* __restrict__ oc, __bf16* __restrict__ ob)
{
    long long i = (long long)blockIdx.x * 256 + threadIdx.x;
    int e = (int)(i & 127);
    int s = (int)((i >> 7) & 511);
    int h = (int)((i >> 16) & 7);
    int b = (int)(i >> 19);
    ob[(((long long)b * S_ + s) * H_ + h) * E_ + e] = (__bf16)oc[i];
}

// Softmax over rows of scores [B,H,S, S]: one wave per row.
__global__ __launch_bounds__(256)
void softmax_k(const float* __restrict__ scores, __bf16* __restrict__ probs,
               const unsigned char* __restrict__ mask, float scale)
{
    const int row  = blockIdx.x * 8 + (threadIdx.x >> 5);  // B*H*S rows
    const int lane = threadIdx.x & 31;
    const float* src = scores + (long long)row * S_;
    __bf16*      dst = probs  + (long long)row * S_;
    const int b = row >> 12;                               // row / (H*S)
    const unsigned char* mrow = mask + b * S_;

    float vals[16];
    float mx = -3.0e38f;
#pragma unroll
    for (int i = 0; i < 16; ++i) {
        const int c = lane + i * 32;
        float v = src[c] * scale;
        if (!mrow[c]) v = -3.0e38f;
        vals[i] = v;
        mx = fmaxf(mx, v);
    }
#pragma unroll
    for (int o = 16; o > 0; o >>= 1) mx = fmaxf(mx, __shfl_xor(mx, o, 32));
    float sum = 0.0f;
#pragma unroll
    for (int i = 0; i < 16; ++i) { float e = __expf(vals[i] - mx); vals[i] = e; sum += e; }
#pragma unroll
    for (int o = 16; o > 0; o >>= 1) sum += __shfl_xor(sum, o, 32);
    const float inv = 1.0f / sum;
#pragma unroll
    for (int i = 0; i < 16; ++i) dst[lane + i * 32] = (__bf16)(vals[i] * inv);
}

// x = LayerNorm(x + addv) * g + b ; also emit bf16 copy. One block per row of 1024.
__global__ __launch_bounds__(256)
void add_ln_k(float* __restrict__ x, const float* __restrict__ addv,
              const float* __restrict__ g, const float* __restrict__ bb,
              __bf16* __restrict__ xb)
{
    __shared__ float red[2][8];
    const int row = blockIdx.x;
    const int tid = threadIdx.x;
    float* xr = x + (long long)row * D_;
    const float* ar = addv + (long long)row * D_;

    float h[4];
    float s = 0.0f;
#pragma unroll
    for (int i = 0; i < 4; ++i) {
        const int c = tid + i * 256;
        h[i] = xr[c] + ar[c];
        s += h[i];
    }
#pragma unroll
    for (int o = 16; o > 0; o >>= 1) s += __shfl_xor(s, o, 32);
    if ((tid & 31) == 0) red[0][tid >> 5] = s;
    __syncthreads();
    float tot = 0.0f;
#pragma unroll
    for (int w = 0; w < 8; ++w) tot += red[0][w];
    const float mu = tot * (1.0f / (float)D_);

    float var = 0.0f;
#pragma unroll
    for (int i = 0; i < 4; ++i) { h[i] -= mu; var += h[i] * h[i]; }
#pragma unroll
    for (int o = 16; o > 0; o >>= 1) var += __shfl_xor(var, o, 32);
    if ((tid & 31) == 0) red[1][tid >> 5] = var;
    __syncthreads();
    float vtot = 0.0f;
#pragma unroll
    for (int w = 0; w < 8; ++w) vtot += red[1][w];
    const float rstd = rsqrtf(vtot * (1.0f / (float)D_) + 1e-5f);

#pragma unroll
    for (int i = 0; i < 4; ++i) {
        const int c = tid + i * 256;
        const float o = h[i] * rstd * g[c] + bb[c];
        xr[c] = o;
        xb[(long long)row * D_ + c] = (__bf16)o;
    }
}

// ---------------------------------------------------------------------------
// Host orchestration
// ---------------------------------------------------------------------------
extern "C" void kernel_launch(void* const* d_in, const int* in_sizes, int n_in,
                              void* d_out, int out_size, void* d_ws, size_t ws_size,
                              hipStream_t stream)
{
    (void)in_sizes; (void)n_in; (void)out_size; (void)ws_size;

    const float* x_in = (const float*)d_in[0];
    const unsigned char* mask = (const unsigned char*)d_in[1];   // jnp bool_
    const float* wq = (const float*)d_in[2];
    const float* wk = (const float*)d_in[3];
    const float* wv = (const float*)d_in[4];
    const float* wo = (const float*)d_in[5];
    const float* bu = (const float*)d_in[6];
    const float* bv = (const float*)d_in[7];
    const float* ln1g = (const float*)d_in[8];
    const float* ln1b = (const float*)d_in[9];
    const float* w1 = (const float*)d_in[10];
    const float* b1 = (const float*)d_in[11];
    const float* w2 = (const float*)d_in[12];
    const float* b2 = (const float*)d_in[13];
    const float* ln2g = (const float*)d_in[14];
    const float* ln2b = (const float*)d_in[15];
    const float* pos  = (const float*)d_in[16];

    char* p = (char*)d_ws;
    auto alloc = [&](size_t bytes) { void* r = (void*)p; p += (bytes + 255) & ~(size_t)255; return r; };

    float*  x_cur = (float*) alloc((size_t)NROWS_ * D_ * 4);
    __bf16* xb    = (__bf16*)alloc((size_t)NROWS_ * D_ * 2);
    __bf16* wqT   = (__bf16*)alloc((size_t)D_ * D_ * 2);
    __bf16* wkT   = (__bf16*)alloc((size_t)D_ * D_ * 2);
    __bf16* wvT   = (__bf16*)alloc((size_t)D_ * D_ * 2);
    __bf16* woT   = (__bf16*)alloc((size_t)D_ * D_ * 2);
    __bf16* w1T   = (__bf16*)alloc((size_t)D_ * FF_ * 2);
    __bf16* w2T   = (__bf16*)alloc((size_t)FF_ * D_ * 2);
    __bf16* posb  = (__bf16*)alloc((size_t)MAXLEN_ * E_ * 2);
    float*  qf    = (float*) alloc((size_t)NROWS_ * D_ * 4);   // reused: ofull
    float*  kf    = (float*) alloc((size_t)NROWS_ * D_ * 4);   // reused: oc
    float*  vf    = (float*) alloc((size_t)NROWS_ * D_ * 4);   // reused: h2
    __bf16* Qbu   = (__bf16*)alloc((size_t)NROWS_ * D_ * 2);   // reused: ob
    __bf16* Qbv   = (__bf16*)alloc((size_t)NROWS_ * D_ * 2);
    __bf16* Kc    = (__bf16*)alloc((size_t)NROWS_ * D_ * 2);
    __bf16* VT    = (__bf16*)alloc((size_t)NROWS_ * D_ * 2);
    float*  scores= (float*) alloc((size_t)B_ * H_ * S_ * S_ * 4); // reused: h1
    __bf16* probs = (__bf16*)alloc((size_t)B_ * H_ * S_ * S_ * 2); // reused: h1b

    float*  ofull = qf;
    float*  oc    = kf;
    float*  h2    = vf;
    __bf16* ob    = Qbu;
    float*  h1    = scores;
    __bf16* h1b   = probs;

    const dim3 blk(256);

    hipMemcpyAsync(x_cur, x_in, (size_t)NROWS_ * D_ * 4, hipMemcpyDeviceToDevice, stream);
    f32_to_bf16_k<<<dim3((NROWS_ * D_) / 256), blk, 0, stream>>>(x_in, xb, (long long)NROWS_ * D_);

    const float scale = 1.0f / sqrtf((float)E_);

    for (int l = 0; l < L_; ++l) {
        // ---- weights -> transposed bf16 operands [N,K]; pos table -> bf16 ----
        transpose_bf16_k<<<dim3((D_ * D_) / 256), blk, 0, stream>>>(wq + (size_t)l * D_ * D_, wqT, D_, D_);
        transpose_bf16_k<<<dim3((D_ * D_) / 256), blk, 0, stream>>>(wk + (size_t)l * D_ * D_, wkT, D_, D_);
        transpose_bf16_k<<<dim3((D_ * D_) / 256), blk, 0, stream>>>(wv + (size_t)l * D_ * D_, wvT, D_, D_);
        transpose_bf16_k<<<dim3((D_ * D_) / 256), blk, 0, stream>>>(wo + (size_t)l * D_ * D_, woT, D_, D_);
        transpose_bf16_k<<<dim3((D_ * FF_) / 256), blk, 0, stream>>>(w1 + (size_t)l * D_ * FF_, w1T, D_, FF_);
        transpose_bf16_k<<<dim3((FF_ * D_) / 256), blk, 0, stream>>>(w2 + (size_t)l * FF_ * D_, w2T, FF_, D_);
        f32_to_bf16_k<<<dim3((MAXLEN_ * E_) / 256), blk, 0, stream>>>(pos + (size_t)l * MAXLEN_ * E_, posb,
                                                                      (long long)MAXLEN_ * E_);

        // ---- Q, K, V projections: [8192,1024] x [1024,1024] ----
        gemm_wmma_bf16<false,false,false><<<dim3(D_/128, NROWS_/128, 1), blk, 0, stream>>>(
            xb, wqT, qf, nullptr, nullptr, D_, D_, D_, D_, 0, 0, 0);
        gemm_wmma_bf16<false,false,false><<<dim3(D_/128, NROWS_/128, 1), blk, 0, stream>>>(
            xb, wkT, kf, nullptr, nullptr, D_, D_, D_, D_, 0, 0, 0);
        gemm_wmma_bf16<false,false,false><<<dim3(D_/128, NROWS_/128, 1), blk, 0, stream>>>(
            xb, wvT, vf, nullptr, nullptr, D_, D_, D_, D_, 0, 0, 0);

        pack_qkv_k<<<dim3((NROWS_ * D_) / 256), blk, 0, stream>>>(
            qf, kf, vf, bu + (size_t)l * H_ * E_, bv + (size_t)l * H_ * E_, Qbu, Qbv, Kc, VT);

        // ---- content scores: z=(b,h), A=[S,E], BT=K [S,E] -> Q.K^T ----
        gemm_wmma_bf16<false,false,false><<<dim3(S_/128, S_/128, B_*H_), blk, 0, stream>>>(
            Qbu, Kc, scores, nullptr, nullptr,
            E_, E_, E_, S_,
            (long long)S_ * E_, (long long)S_ * E_, (long long)S_ * S_);

        // ---- position scores: z=q; BT rows are pos_table rows (no transpose):
        // BT[n][e] = pos[n + offset - q][e]; base advances by -E per q. Accumulate.
        gemm_wmma_bf16<true,false,false><<<dim3(S_/128, 1, S_), blk, 0, stream>>>(
            Qbv, posb + (size_t)(MAXLEN_/2) * E_, scores, nullptr, nullptr,
            E_, S_ * E_, E_, S_ * S_,
            (long long)E_, (long long)(-E_), (long long)S_);

        softmax_k<<<dim3(B_ * H_ * S_ / 8), blk, 0, stream>>>(scores, probs, mask, scale);

        // ---- A @ V: z=(b,h), A=probs [S,S], BT=V^T [E,S] -> oc [B,H,S,E] ----
        gemm_wmma_bf16<false,false,false><<<dim3(E_/128, S_/128, B_*H_), blk, 0, stream>>>(
            probs, VT, oc, nullptr, nullptr,
            S_, S_, S_, E_,
            (long long)S_ * S_, (long long)E_ * S_, (long long)S_ * E_);

        permute_o_k<<<dim3((NROWS_ * D_) / 256), blk, 0, stream>>>(oc, ob);
        gemm_wmma_bf16<false,false,false><<<dim3(D_/128, NROWS_/128, 1), blk, 0, stream>>>(
            ob, woT, ofull, nullptr, nullptr, D_, D_, D_, D_, 0, 0, 0);

        add_ln_k<<<dim3(NROWS_), blk, 0, stream>>>(
            x_cur, ofull, ln1g + (size_t)l * D_, ln1b + (size_t)l * D_, xb);

        // ---- FFN ----
        gemm_wmma_bf16<false,true,true><<<dim3(FF_/128, NROWS_/128, 1), blk, 0, stream>>>(
            xb, w1T, h1, h1b, b1 + (size_t)l * FF_, D_, D_, D_, FF_, 0, 0, 0);
        gemm_wmma_bf16<false,true,false><<<dim3(D_/128, NROWS_/128, 1), blk, 0, stream>>>(
            h1b, w2T, h2, nullptr, b2 + (size_t)l * D_, FF_, FF_, FF_, D_, 0, 0, 0);

        add_ln_k<<<dim3(NROWS_), blk, 0, stream>>>(
            x_cur, h2, ln2g + (size_t)l * D_, ln2b + (size_t)l * D_, xb);
    }

    hipMemcpyAsync(d_out, x_cur, (size_t)NROWS_ * D_ * 4, hipMemcpyDeviceToDevice, stream);
}
